// EGAT_35442070126741
// MI455X (gfx1250) — compile-verified, matrix operand order checked by reading, not run on hardware
//
#include <hip/hip_runtime.h>
#include <hip/hip_bf16.h>
#include <math.h>

#define NNODE 50000
#define NEDGE 800000
#define FDIM  128
#define HEADS 4
#define N_OUT 10
#define NEG_SLOPE 0.01f

typedef __bf16 bf16;
typedef __attribute__((ext_vector_type(16))) __bf16 v16bf;
typedef __attribute__((ext_vector_type(8)))  __bf16 v8bf;
typedef __attribute__((ext_vector_type(8)))  float  v8f;

// ---- monotone float<->int mapping so integer atomicMax gives float max ----
__device__ __forceinline__ int f2oi(float f) {
    int i = __float_as_int(f);
    return i >= 0 ? i : (i ^ 0x7fffffff);
}
__device__ __forceinline__ float oi2f(int i) {
    return __int_as_float(i >= 0 ? i : (i ^ 0x7fffffff));
}

// -------------------- fills --------------------
__global__ void fill_f32(float* p, float v, int n) {
    int i = blockIdx.x * blockDim.x + threadIdx.x;
    if (i < n) p[i] = v;
}
__global__ void fill_i32(int* p, int v, int n) {
    int i = blockIdx.x * blockDim.x + threadIdx.x;
    if (i < n) p[i] = v;
}

// -------------------- small projections: f_ni = X@W_ni, f_nj = X@W_nj ----
// one wave (32 lanes) per node; lane loads float4, cross-lane reduce (wave32)
__global__ void __launch_bounds__(256)
proj_ninj(const float* __restrict__ X, const float* __restrict__ Wni,
          const float* __restrict__ Wnj, float* __restrict__ fni,
          float* __restrict__ fnj, int nrows) {
    int node = (blockIdx.x * blockDim.x + threadIdx.x) >> 5;
    int lane = threadIdx.x & 31;
    if (node >= nrows) return;
    float4 x = *(const float4*)(X + (size_t)node * FDIM + lane * 4);
    float xv[4] = {x.x, x.y, x.z, x.w};
    float sni[HEADS] = {0.f, 0.f, 0.f, 0.f};
    float snj[HEADS] = {0.f, 0.f, 0.f, 0.f};
#pragma unroll
    for (int i = 0; i < 4; ++i) {
        int k = lane * 4 + i;
#pragma unroll
        for (int h = 0; h < HEADS; ++h) {
            sni[h] += xv[i] * Wni[k * HEADS + h];
            snj[h] += xv[i] * Wnj[k * HEADS + h];
        }
    }
#pragma unroll
    for (int h = 0; h < HEADS; ++h) {
#pragma unroll
        for (int off = 16; off > 0; off >>= 1) {
            sni[h] += __shfl_xor(sni[h], off, 32);
            snj[h] += __shfl_xor(snj[h], off, 32);
        }
    }
    if (lane == 0) {
#pragma unroll
        for (int h = 0; h < HEADS; ++h) {
            fni[(size_t)node * HEADS + h] = sni[h];
            fnj[(size_t)node * HEADS + h] = snj[h];
        }
    }
}

// -------------------- Y[N,128] = X[N,128] @ W[128,128] via bf16 WMMA ------
// 256 threads = 8 waves; each wave owns a 16x128 output strip.
// LDS: X tile (bf16) + W^T (bf16), row stride 136 (16B-aligned, padded).
// All 8 B fragments of a K-step are loaded BEFORE the 8-WMMA chain so the
// scheduler can batch ds_load_b128 issues and avoid a full dscnt wait per WMMA.
__global__ void __launch_bounds__(256)
gemm128_bf16(const float* __restrict__ X, const float* __restrict__ W,
             float* __restrict__ Y, int nrows) {
    extern __shared__ char smem_raw[];
    bf16(*sA)[136] = (bf16(*)[136])smem_raw;                              // X tile
    bf16(*sB)[136] = (bf16(*)[136])(smem_raw + 128 * 136 * sizeof(bf16)); // W^T [n][k]

    const int tid = threadIdx.x;
    const int rowbase = blockIdx.x * 128;

    // stage W^T: sB[n][k] = W[k*128+n]; idx = k*128+n -> coalesced global read
    for (int idx = tid; idx < 128 * 128; idx += 256) {
        int k = idx >> 7, n = idx & 127;
        sB[n][k] = (bf16)W[idx];
    }
    // stage X tile (zero-pad past nrows)
    for (int idx = tid; idx < 128 * 128; idx += 256) {
        int r = idx >> 7, c = idx & 127;
        int gr = rowbase + r;
        float v = (gr < nrows) ? X[(size_t)gr * FDIM + c] : 0.0f;
        sA[r][c] = (bf16)v;
    }
    __syncthreads();

    const int wave = tid >> 5;
    const int lane = tid & 31;
    const int mrow = wave * 16 + (lane & 15);
    const int kshift = (lane >> 4) ? 8 : 0;  // ISA 16-bit A layout: hi lanes hold K+8

    union frag { v16bf v; v8bf h[2]; };

    v8f acc[8] = {};
#pragma unroll
    for (int kb = 0; kb < 128; kb += 32) {
        frag au;
        au.h[0] = *(const v8bf*)&sA[mrow][kb + kshift];
        au.h[1] = *(const v8bf*)&sA[mrow][kb + kshift + 16];
        frag bu[8];
#pragma unroll
        for (int c = 0; c < 8; ++c) {
            const int ncol = c * 16 + (lane & 15);
            bu[c].h[0] = *(const v8bf*)&sB[ncol][kb + kshift];
            bu[c].h[1] = *(const v8bf*)&sB[ncol][kb + kshift + 16];
        }
#pragma unroll
        for (int c = 0; c < 8; ++c) {
            acc[c] = __builtin_amdgcn_wmma_f32_16x16x32_bf16(
                false, au.v, false, bu[c].v, (short)0, acc[c], false, false);
        }
    }

    // D layout: VGPR r -> M=r (lanes 0-15), M=r+8 (lanes 16-31); N = lane&15
    const int rowsel = (lane >> 4) ? 8 : 0;
#pragma unroll
    for (int c = 0; c < 8; ++c) {
#pragma unroll
        for (int r = 0; r < 8; ++r) {
            int gr = rowbase + wave * 16 + rowsel + r;
            if (gr < nrows) Y[(size_t)gr * FDIM + c * 16 + (lane & 15)] = acc[c][r];
        }
    }
}

// -------------------- per-edge attention logits + segment max -------------
__global__ void edge_logits(const int* __restrict__ src, const int* __restrict__ dst,
                            const float* __restrict__ efeat,
                            const float* __restrict__ fni, const float* __restrict__ fnj,
                            const float* __restrict__ Wfij, const float* __restrict__ attn,
                            const float* __restrict__ be,
                            float* __restrict__ ebuf, int* __restrict__ emax) {
    int e = blockIdx.x * blockDim.x + threadIdx.x;
    if (e >= NEDGE) return;
    int s = src[e], d = dst[e];
    float ef = efeat[e];
    float4 ni = *(const float4*)(fni + (size_t)s * HEADS);
    float4 nj = *(const float4*)(fnj + (size_t)d * HEADS);
    float nis[4] = {ni.x, ni.y, ni.z, ni.w};
    float njs[4] = {nj.x, nj.y, nj.z, nj.w};
    float ev[4];
#pragma unroll
    for (int h = 0; h < HEADS; ++h) {
        float f = nis[h] + njs[h] + ef * Wfij[h] + be[h];
        f = (f > 0.f) ? f : NEG_SLOPE * f;   // leaky relu
        float eh = f * attn[h];              // FE == 1
        ev[h] = eh;
        atomicMax(&emax[(size_t)d * HEADS + h], f2oi(eh));
    }
    *(float4*)(ebuf + (size_t)e * HEADS) = make_float4(ev[0], ev[1], ev[2], ev[3]);
}

// -------------------- exp(e - max) + segment sum --------------------------
__global__ void edge_expsum(const int* __restrict__ dst, float* __restrict__ ebuf,
                            const int* __restrict__ emax, float* __restrict__ esum) {
    int e = blockIdx.x * blockDim.x + threadIdx.x;
    if (e >= NEDGE) return;
    int d = dst[e];
    float4 ev4 = *(const float4*)(ebuf + (size_t)e * HEADS);
    float ev[4] = {ev4.x, ev4.y, ev4.z, ev4.w};
#pragma unroll
    for (int h = 0; h < HEADS; ++h) {
        float m = oi2f(emax[(size_t)d * HEADS + h]);
        float ee = expf(ev[h] - m);
        ev[h] = ee;
        atomicAdd(&esum[(size_t)d * HEADS + h], ee);
    }
    *(float4*)(ebuf + (size_t)e * HEADS) = make_float4(ev[0], ev[1], ev[2], ev[3]);
}

// -------------------- m = h[src]*alpha, segment-sum into rst[dst] ---------
// one wave32 per edge: coalesced 512B row gather, 4 f32 atomics per lane (L2)
__global__ void __launch_bounds__(256)
edge_scatter(const int* __restrict__ src, const int* __restrict__ dst,
             const float* __restrict__ hbuf, const float* __restrict__ ebuf,
             const float* __restrict__ esum, float* __restrict__ rst) {
    int e = (blockIdx.x * blockDim.x + threadIdx.x) >> 5;
    int lane = threadIdx.x & 31;
    if (e >= NEDGE) return;
    int s = src[e], d = dst[e];
    int head = lane >> 3;  // channels [4*lane,4*lane+3] all in head lane/8
    float alpha = ebuf[(size_t)e * HEADS + head] / esum[(size_t)d * HEADS + head];
    int c = lane * 4;
    float4 hv = *(const float4*)(hbuf + (size_t)s * FDIM + c);
    float* rp = rst + (size_t)d * FDIM + c;
    atomicAdd(rp + 0, hv.x * alpha);
    atomicAdd(rp + 1, hv.y * alpha);
    atomicAdd(rp + 2, hv.z * alpha);
    atomicAdd(rp + 3, hv.w * alpha);
}

// -------------------- relu copy ------------------------------------------
__global__ void relu_copy(const float* __restrict__ in, float* __restrict__ out, int n) {
    int i = blockIdx.x * blockDim.x + threadIdx.x;
    if (i < n) out[i] = fmaxf(in[i], 0.0f);
}

// -------------------- column sums for mean pooling ------------------------
__global__ void colsum(const float* __restrict__ rst, float* __restrict__ pooled) {
    int c = threadIdx.x;  // 128 threads
    float s = 0.0f;
    for (int n = blockIdx.x; n < NNODE; n += gridDim.x)
        s += rst[(size_t)n * FDIM + c];
    atomicAdd(&pooled[c], s);
}

// -------------------- final FC: out = mean @ W_fc + b ---------------------
__global__ void final_fc(const float* __restrict__ pooled, const float* __restrict__ Wfc,
                         const float* __restrict__ bfc, float* __restrict__ out) {
    int j = threadIdx.x;
    if (j >= N_OUT) return;
    float s = 0.0f;
    for (int c = 0; c < FDIM; ++c) s += pooled[c] * Wfc[c * N_OUT + j];
    out[j] = s * (1.0f / (float)NNODE) + bfc[j];
}

extern "C" void kernel_launch(void* const* d_in, const int* in_sizes, int n_in,
                              void* d_out, int out_size, void* d_ws, size_t ws_size,
                              hipStream_t stream) {
    (void)in_sizes; (void)n_in; (void)out_size; (void)ws_size;
    const float* X    = (const float*)d_in[0];
    const float* EF   = (const float*)d_in[1];
    const int*   src  = (const int*)d_in[2];
    const int*   dst  = (const int*)d_in[3];
    const float* Wn1  = (const float*)d_in[4];
    const float* Wni1 = (const float*)d_in[5];
    const float* Wnj1 = (const float*)d_in[6];
    const float* Wf1  = (const float*)d_in[7];
    const float* at1  = (const float*)d_in[8];
    const float* be1  = (const float*)d_in[9];
    const float* Wn2  = (const float*)d_in[10];
    const float* Wni2 = (const float*)d_in[11];
    const float* Wnj2 = (const float*)d_in[12];
    const float* Wf2  = (const float*)d_in[13];
    const float* at2  = (const float*)d_in[14];
    const float* be2  = (const float*)d_in[15];
    const float* Wfc  = (const float*)d_in[16];
    const float* bfc  = (const float*)d_in[17];
    float* out = (float*)d_out;

    // workspace layout (floats)
    float* ws = (float*)d_ws;
    size_t off = 0;
    float* fni    = ws + off; off += (size_t)NNODE * HEADS;
    float* fnj    = ws + off; off += (size_t)NNODE * HEADS;
    int*   emax   = (int*)(ws + off); off += (size_t)NNODE * HEADS;
    float* esum   = ws + off; off += (size_t)NNODE * HEADS;
    float* ebuf   = ws + off; off += (size_t)NEDGE * HEADS;
    float* hbuf   = ws + off; off += (size_t)NNODE * FDIM;
    float* rstb   = ws + off; off += (size_t)NNODE * FDIM;
    float* x2     = ws + off; off += (size_t)NNODE * FDIM;
    float* pooled = ws + off; off += FDIM;

    const int NH  = NNODE * HEADS;   // 200000
    const int NF  = NNODE * FDIM;    // 6.4M
    const int OI_NEG_INF = (int)0x807FFFFF;  // f2oi(-inf)
    const int gemm_grid  = (NNODE + 127) / 128;          // 391
    const size_t gemm_lds = 2u * 128u * 136u * sizeof(bf16); // 69632 B (<320KB/WGP)

    auto run_layer = [&](const float* Xin, const float* Wn, const float* Wni,
                         const float* Wnj, const float* Wf, const float* at,
                         const float* be) {
        fill_i32<<<(NH + 255) / 256, 256, 0, stream>>>(emax, OI_NEG_INF, NH);
        fill_f32<<<(NH + 255) / 256, 256, 0, stream>>>(esum, 0.0f, NH);
        fill_f32<<<(NF + 255) / 256, 256, 0, stream>>>(rstb, 0.0f, NF);
        proj_ninj<<<(NNODE + 7) / 8, 256, 0, stream>>>(Xin, Wni, Wnj, fni, fnj, NNODE);
        gemm128_bf16<<<gemm_grid, 256, gemm_lds, stream>>>(Xin, Wn, hbuf, NNODE);
        edge_logits<<<(NEDGE + 255) / 256, 256, 0, stream>>>(src, dst, EF, fni, fnj,
                                                             Wf, at, be, ebuf, emax);
        edge_expsum<<<(NEDGE + 255) / 256, 256, 0, stream>>>(dst, ebuf, emax, esum);
        edge_scatter<<<NEDGE / 8, 256, 0, stream>>>(src, dst, hbuf, ebuf, esum, rstb);
    };

    // layer 1
    run_layer(X, Wn1, Wni1, Wnj1, Wf1, at1, be1);
    relu_copy<<<(NF + 255) / 256, 256, 0, stream>>>(rstb, x2, NF);
    // layer 2
    run_layer(x2, Wn2, Wni2, Wnj2, Wf2, at2, be2);
    // mean pool + FC
    fill_f32<<<1, 128, 0, stream>>>(pooled, 0.0f, FDIM);
    colsum<<<512, 128, 0, stream>>>(rstb, pooled);
    final_fc<<<1, 32, 0, stream>>>(pooled, Wfc, bfc, out);
}
